// EncoderBiRNN_42013370089576
// MI455X (gfx1250) — matrix operand order repaired and checked by reading, not compiled, not run on hardware
//
#include <hip/hip_runtime.h>

typedef unsigned short u16;
typedef unsigned int   u32;
typedef __attribute__((ext_vector_type(16))) __bf16 v16bf;
typedef __attribute__((ext_vector_type(8)))  float  v8f;

#define T_SEQ 512
#define BATCH 64
#define EMBD  512
#define H2D   512
#define G3    1536   // 3*H2
#define NTOT  3072   // both directions
#define MTOT  (T_SEQ*BATCH)
#define NBLK  16     // blocks per direction in the scan

__device__ __forceinline__ u16 f2bf(float f) {
  u32 u = __float_as_uint(f);
  u32 r = (u + 0x7FFFu + ((u >> 16) & 1u)) >> 16;
  return (u16)r;
}
__device__ __forceinline__ float bf2f(u16 h) {
  return __uint_as_float(((u32)h) << 16);
}
__device__ __forceinline__ float sigm(float x) {
  return 1.0f / (1.0f + __expf(-x));
}

union Frag {
  uint4 q[2];
  v16bf v;
};

// ---------------------------------------------------------------------------
// Kernel 0: zero the per-step barrier counters and the h0 buffer.
// Runs every launch -> no state leaks across graph replays.
// ---------------------------------------------------------------------------
__global__ void init_state(int* __restrict__ counters,   // [2][T_SEQ]
                           u16* __restrict__ hbuf) {     // [2][2][64][512]
  const int NC = 2 * T_SEQ;
  const int NH = 2 * BATCH * H2D;   // buffer 0, both directions
  for (int i = blockIdx.x * blockDim.x + threadIdx.x; i < NC + NH;
       i += gridDim.x * blockDim.x) {
    if (i < NC) counters[i] = 0;
    else        hbuf[i - NC] = 0;
  }
}

// ---------------------------------------------------------------------------
// Kernel 1: convert f32 weights -> bf16.
//   wih_bf[3072][512]  (rows 0..1535 = w_ih_f, 1536..3071 = w_ih_r)
//   whh_bf[2][1536][512]
// ---------------------------------------------------------------------------
__global__ void prep_weights(const float* __restrict__ wih_f,
                             const float* __restrict__ wih_r,
                             const float* __restrict__ whh_f,
                             const float* __restrict__ whh_r,
                             u16* __restrict__ wih_bf,
                             u16* __restrict__ whh_bf) {
  const int HALF = G3 * EMBD;          // 786432
  const int N    = 2 * HALF;
  for (int i = blockIdx.x * blockDim.x + threadIdx.x; i < N;
       i += gridDim.x * blockDim.x) {
    float a = (i < HALF) ? wih_f[i] : wih_r[i - HALF];
    float b = (i < HALF) ? whh_f[i] : whh_r[i - HALF];
    wih_bf[i] = f2bf(a);
    whh_bf[i] = f2bf(b);
  }
}

// ---------------------------------------------------------------------------
// Kernel 2: fused embedding-gather + input projection GEMM.
//   C[m][n] = sum_k emb[tokens[m]][k] * wih[n][k] + b_ih[n]
//   m = t*64 + b (M = 32768), n in [0,3072). Output xproj bf16 [T][B][3072].
// Block = 256 threads (8 waves), tile 128x128, K-step 32.
// ---------------------------------------------------------------------------
__global__ __launch_bounds__(256) void xproj_gemm(
    const int*   __restrict__ tokens,     // [64][512]
    const float* __restrict__ emb,        // [32000][512]
    const u16*   __restrict__ wih,        // [3072][512] bf16
    const float* __restrict__ bih_f,      // [1536]
    const float* __restrict__ bih_r,      // [1536]
    u16*         __restrict__ xproj) {    // [32768][3072] bf16
  __shared__ u16 sA[128 * 40];            // 128 rows x 32 k (+8 pad), bf16

  const int m0   = blockIdx.x * 128;
  const int n0   = blockIdx.y * 128;
  const int tid  = threadIdx.x;
  const int lane = tid & 31;
  const int wid  = tid >> 5;
  const int wm   = wid & 3;               // 4 wave rows  -> 32 M each
  const int wn   = wid >> 2;              // 2 wave cols  -> 64 N each
  const int half = lane >> 4;
  const int l15  = lane & 15;

  const int arow = tid >> 1;              // 0..127
  const int aseg = (tid & 1) * 16;        // 0 or 16
  const int mg   = m0 + arow;
  const int tok  = tokens[(mg & 63) * T_SEQ + (mg >> 6)];
  const float* __restrict__ arow_ptr = emb + (size_t)tok * EMBD;

  v8f acc[2][4];
#pragma unroll
  for (int i = 0; i < 2; ++i)
#pragma unroll
    for (int j = 0; j < 4; ++j) acc[i][j] = (v8f)0.0f;

  for (int kk = 0; kk < EMBD / 32; ++kk) {
    __syncthreads();
    {
      const float* src = arow_ptr + kk * 32 + aseg;
      u32* dst = (u32*)&sA[arow * 40 + aseg];
#pragma unroll
      for (int q = 0; q < 4; ++q) {
        float4 f = *(const float4*)(src + q * 4);
        dst[2 * q + 0] = (u32)f2bf(f.x) | ((u32)f2bf(f.y) << 16);
        dst[2 * q + 1] = (u32)f2bf(f.z) | ((u32)f2bf(f.w) << 16);
      }
    }
    __syncthreads();

    Frag a[2], b[4];
#pragma unroll
    for (int im = 0; im < 2; ++im) {
      const uint4* p = (const uint4*)&sA[(wm * 32 + im * 16 + l15) * 40];
      a[im].q[0] = p[2 * half];
      a[im].q[1] = p[2 * half + 1];
    }
#pragma unroll
    for (int in = 0; in < 4; ++in) {
      const int n = n0 + wn * 64 + in * 16 + l15;
      const uint4* p =
          (const uint4*)(wih + (size_t)n * EMBD + kk * 32 + half * 16);
      b[in].q[0] = p[0];
      b[in].q[1] = p[1];
    }
#pragma unroll
    for (int im = 0; im < 2; ++im)
#pragma unroll
      for (int in = 0; in < 4; ++in)
        acc[im][in] = __builtin_amdgcn_wmma_f32_16x16x32_bf16(
            false, a[im].v, false, b[in].v, (short)0, acc[im][in], false, false);
  }

#pragma unroll
  for (int im = 0; im < 2; ++im) {
    const int fm0 = m0 + wm * 32 + im * 16;
#pragma unroll
    for (int in = 0; in < 4; ++in) {
      const int n   = n0 + wn * 64 + in * 16 + l15;
      const float bias = (n < G3) ? bih_f[n] : bih_r[n - G3];
#pragma unroll
      for (int i = 0; i < 8; ++i) {
        const int mrow = fm0 + i + half * 8;
        xproj[(size_t)mrow * NTOT + n] = f2bf(acc[im][in][i] + bias);
      }
    }
  }
}

// ---------------------------------------------------------------------------
// Kernel 3: GRU scan spread over 16 WGPs per direction.
// Grid = 32 blocks: blockIdx >> 4 = direction, blockIdx & 15 = 32-column
// j-slice. Block = 256 threads (8 waves): wave = (m-tile 0..3, j-tile 0..1),
// 3 gate accumulators per wave. Hidden state double-buffered in global
// (L2-resident, 128 KB); per-step sync = per-step global atomic counter.
// ---------------------------------------------------------------------------
__global__ __launch_bounds__(256) void gru_scan(
    const u16*   __restrict__ whh,       // [2][1536][512] bf16
    const float* __restrict__ bhh_f,     // [1536]
    const float* __restrict__ bhh_r,     // [1536]
    const u16*   __restrict__ xproj,     // [T][B][3072] bf16
    u16*         __restrict__ hbuf,      // [2][2][64][512] bf16
    int*         __restrict__ counters,  // [2][T_SEQ]
    float*       __restrict__ out) {     // [1][64][1024]
  const int dir   = blockIdx.x >> 4;
  const int slice = blockIdx.x & 15;

  const u16* __restrict__ whh_d = whh + (size_t)dir * G3 * H2D;
  const float* __restrict__ bhh = dir ? bhh_r : bhh_f;
  int* __restrict__ ctr_base = counters + dir * T_SEQ;

  const int tid  = threadIdx.x;
  const int lane = tid & 31;
  const int wid  = tid >> 5;            // 0..7
  const int m    = wid & 3;             // M-tile (16 rows of batch)
  const int jt   = wid >> 2;            // j-tile within slice (0..1)
  const int half = lane >> 4;
  const int l15  = lane & 15;
  const int j    = slice * 32 + jt * 16 + l15;   // output column [0,512)

  // Per-gate B row pointers: W_hh row (g*512 + j), k-offset half*16
  const u16* bptr0 = whh_d + (size_t)(0 * H2D + j) * H2D + half * 16;
  const u16* bptr1 = whh_d + (size_t)(1 * H2D + j) * H2D + half * 16;
  const u16* bptr2 = whh_d + (size_t)(2 * H2D + j) * H2D + half * 16;
  const float bh0 = bhh[j], bh1 = bhh[H2D + j], bh2 = bhh[2 * H2D + j];

  int cur = 0;
  for (int t = 0; t < T_SEQ; ++t) {
    const u16* __restrict__ h_cur = hbuf + (size_t)(cur * 2 + dir) * BATCH * H2D;
    u16* __restrict__ h_nxt = hbuf + (size_t)((cur ^ 1) * 2 + dir) * BATCH * H2D;
    const int tt = dir ? (T_SEQ - 1 - t) : t;
    const u16* __restrict__ xp = xproj + (size_t)tt * BATCH * NTOT + dir * G3;

    v8f acc0 = (v8f)0.0f, acc1 = (v8f)0.0f, acc2 = (v8f)0.0f;

    for (int kk = 0; kk < H2D / 32; ++kk) {
      Frag bf0, bf1, bf2, af;
      {
        const uint4* p0 = (const uint4*)(bptr0 + kk * 32);
        const uint4* p1 = (const uint4*)(bptr1 + kk * 32);
        const uint4* p2 = (const uint4*)(bptr2 + kk * 32);
        bf0.q[0] = p0[0]; bf0.q[1] = p0[1];
        bf1.q[0] = p1[0]; bf1.q[1] = p1[1];
        bf2.q[0] = p2[0]; bf2.q[1] = p2[1];
        const int brow = m * 16 + l15;
        const uint4* pa =
            (const uint4*)(h_cur + (size_t)brow * H2D + kk * 32 + half * 16);
        af.q[0] = pa[0];
        af.q[1] = pa[1];   // +8 u16
      }
      acc0 = __builtin_amdgcn_wmma_f32_16x16x32_bf16(
          false, af.v, false, bf0.v, (short)0, acc0, false, false);
      acc1 = __builtin_amdgcn_wmma_f32_16x16x32_bf16(
          false, af.v, false, bf1.v, (short)0, acc1, false, false);
      acc2 = __builtin_amdgcn_wmma_f32_16x16x32_bf16(
          false, af.v, false, bf2.v, (short)0, acc2, false, false);
    }

    // gate math in accumulator layout
#pragma unroll
    for (int i = 0; i < 8; ++i) {
      const int brow = m * 16 + i + half * 8;
      const size_t xb = (size_t)brow * NTOT;
      const float xr = bf2f(xp[xb + j]);
      const float xz = bf2f(xp[xb + H2D + j]);
      const float xn = bf2f(xp[xb + 2 * H2D + j]);
      const float ho = bf2f(h_cur[(size_t)brow * H2D + j]);
      const float r  = sigm(xr + acc0[i] + bh0);
      const float z  = sigm(xz + acc1[i] + bh1);
      const float nn = tanhf(xn + r * (acc2[i] + bh2));
      const float hn = (1.0f - z) * nn + z * ho;
      h_nxt[(size_t)brow * H2D + j] = f2bf(hn);
      if (t == T_SEQ - 1) out[brow * 1024 + dir * H2D + j] = hn;
    }

    if (t < T_SEQ - 1) {
      // inter-WGP step barrier: release our slice, wait for all 16 blocks
      __threadfence();
      __syncthreads();
      int* ctr = ctr_base + t;
      if (tid == 0) {
        __hip_atomic_fetch_add(ctr, 1, __ATOMIC_RELEASE,
                               __HIP_MEMORY_SCOPE_AGENT);
        while (__hip_atomic_load(ctr, __ATOMIC_ACQUIRE,
                                 __HIP_MEMORY_SCOPE_AGENT) < NBLK) {
          __builtin_amdgcn_s_sleep(2);
        }
      }
      __syncthreads();
      __threadfence();
    }
    cur ^= 1;
  }
}

// ---------------------------------------------------------------------------
extern "C" void kernel_launch(void* const* d_in, const int* in_sizes, int n_in,
                              void* d_out, int out_size, void* d_ws,
                              size_t ws_size, hipStream_t stream) {
  const int*   tokens = (const int*)d_in[0];
  const float* emb    = (const float*)d_in[1];
  const float* wih_f  = (const float*)d_in[2];
  const float* whh_f  = (const float*)d_in[3];
  const float* bih_f  = (const float*)d_in[4];
  const float* bhh_f  = (const float*)d_in[5];
  const float* wih_r  = (const float*)d_in[6];
  const float* whh_r  = (const float*)d_in[7];
  const float* bih_r  = (const float*)d_in[8];
  const float* bhh_r  = (const float*)d_in[9];
  float* out = (float*)d_out;

  // workspace layout
  int* counters = (int*)d_ws;                              // 2*512 ints (4 KB)
  u16* hbuf     = (u16*)((char*)d_ws + 4096);              // 2*2*64*512 (256 KB)
  u16* wih_bf   = hbuf + (size_t)2 * 2 * BATCH * H2D;      // 3072*512
  u16* whh_bf   = wih_bf + (size_t)NTOT * EMBD;            // 2*1536*512
  u16* xproj    = whh_bf + (size_t)2 * G3 * H2D;           // 32768*3072

  init_state<<<68, 256, 0, stream>>>(counters, hbuf);
  prep_weights<<<3072, 256, 0, stream>>>(wih_f, wih_r, whh_f, whh_r,
                                         wih_bf, whh_bf);

  dim3 ggrid(MTOT / 128, NTOT / 128);                      // 256 x 24
  xproj_gemm<<<ggrid, 256, 0, stream>>>(tokens, emb, wih_bf, bih_f, bih_r,
                                        xproj);

  gru_scan<<<2 * NBLK, 256, 0, stream>>>(whh_bf, bhh_f, bhh_r, xproj,
                                         hbuf, counters, out);
}
